// SLAHead_16896401343238
// MI455X (gfx1250) — matrix operand order repaired and checked by reading, not compiled, not run on hardware
//
#include <hip/hip_runtime.h>
#include <hip/hip_bf16.h>
#include <math.h>

// ---------------------------------------------------------------------------
// SLAHead decoder for MI455X (gfx1250, wave32, WMMA).
//   Kernel 1: tiled transpose fea[B,C,N] -> featT f32 [B,N,C] + featB bf16
//   Kernel 2: i2h_w f32 -> bf16
//   Kernel 3: h_proj[B,N,HID] = featB @ i2hB^T via v_wmma_f32_16x16x32_bf16
//             (pure bf16 b128 fragment loads, no per-step cvt, unrolled x4)
//   Kernel 4: 201-step GRU-attention decode, 1 workgroup per batch element
// ---------------------------------------------------------------------------

#define BSZ   16
#define CCH   512
#define NPOS  400     // 20*20
#define HID   256
#define NEMB  30
#define LOCN  8
#define TSTEPS 201
#define XDIM  (CCH + NEMB)   // 542
#define OUTW  (NEMB + LOCN)  // 38

typedef __attribute__((ext_vector_type(16))) __bf16 v16bf;
typedef __attribute__((ext_vector_type(8)))  __bf16 v8bf;
typedef __attribute__((ext_vector_type(8)))  float  v8f;

// ---------------------------------------------------------------------------
// Kernel 1: featT[b][n][c] = fea[b][c][n] (f32) and featB (bf16 copy).
// 16x16 LDS tiles, coalesced both sides. grid (25, 32, 16), block 256.
// ---------------------------------------------------------------------------
__global__ void sla_transpose_kernel(const float* __restrict__ fea,
                                     float* __restrict__ featT,
                                     __bf16* __restrict__ featB) {
    __shared__ float tile[16][17];
    const int b  = blockIdx.z;
    const int n0 = blockIdx.x * 16;   // 400/16 = 25
    const int c0 = blockIdx.y * 16;   // 512/16 = 32
    const int tx = threadIdx.x & 15;
    const int ty = threadIdx.x >> 4;
    tile[ty][tx] = fea[((size_t)b * CCH + (c0 + ty)) * NPOS + (n0 + tx)];
    __syncthreads();
    const float v = tile[tx][ty];
    const size_t o = ((size_t)b * NPOS + (n0 + ty)) * CCH + (c0 + tx);
    featT[o] = v;
    featB[o] = (__bf16)v;
}

// ---------------------------------------------------------------------------
// Kernel 2: i2hB = bf16(i2h_w). 256*512 = 131072 elems. grid 128, block 256,
// 4 elems/thread.
// ---------------------------------------------------------------------------
__global__ void sla_cvt_w_kernel(const float* __restrict__ w,
                                 __bf16* __restrict__ wb) {
    const int i = (blockIdx.x * 256 + threadIdx.x) * 4;
    const float4 v = *(const float4*)(w + i);
    wb[i + 0] = (__bf16)v.x;
    wb[i + 1] = (__bf16)v.y;
    wb[i + 2] = (__bf16)v.z;
    wb[i + 3] = (__bf16)v.w;
}

// ---------------------------------------------------------------------------
// Kernel 3: h_proj = featB @ i2hB^T, bf16 WMMA, f32 accumulate.
//   M = B*N = 6400 rows (400 per batch, 16 | 400), Nout = HID = 256, K = 512.
//   One wave per 16x16 tile; grid (400 m-tiles, 4 h-groups), block 128.
// A fragment (16-bit 16x32, ISA 7.12.2): lane l -> M = l%16;
//   elems 0..7  = K[k0 + (l<16?0:8)   + 0..7]   (8 contiguous bf16 = b128)
//   elems 8..15 = K[k0 + 16 + (l<16?0:8) + 0..7] (b128)
// B fragment (32x16): lane l -> Nout = l%16; same K mapping as A
//   (elems 0..7 = K[k0+(l<16?0:8)+0..7], elems 8..15 = K[k0+16+(l<16?0:8)+0..7])
// C/D: lane l, elem r -> (M = r + (l<16?0:8), Nout = l%16)
// ---------------------------------------------------------------------------
__global__ void sla_hproj_wmma_kernel(const __bf16* __restrict__ featB,
                                      const __bf16* __restrict__ i2hB,
                                      float* __restrict__ hproj) {
    const int lane  = threadIdx.x & 31;
    const int wv    = threadIdx.x >> 5;
    const int m0    = blockIdx.x * 16;          // row tile (never crosses b)
    const int h0    = (blockIdx.y * 4 + wv) * 16;
    const int half  = lane >> 4;                // 0: lanes 0-15, 1: lanes 16-31
    const int l16   = lane & 15;

    const __bf16* arow = featB + (size_t)(m0 + l16) * CCH + half * 8;
    const __bf16* brow = i2hB  + (size_t)(h0 + l16) * CCH + half * 8;

    v8f acc = {0.f, 0.f, 0.f, 0.f, 0.f, 0.f, 0.f, 0.f};

#pragma unroll 4
    for (int k0 = 0; k0 < CCH; k0 += 32) {
        const v8bf a_lo = *(const v8bf*)(arow + k0);        // 16B aligned
        const v8bf a_hi = *(const v8bf*)(arow + k0 + 16);
        const v8bf b_lo = *(const v8bf*)(brow + k0);
        const v8bf b_hi = *(const v8bf*)(brow + k0 + 16);
        const v16bf a  = __builtin_shufflevector(a_lo, a_hi,
            0, 1, 2, 3, 4, 5, 6, 7, 8, 9, 10, 11, 12, 13, 14, 15);
        const v16bf bm = __builtin_shufflevector(b_lo, b_hi,
            0, 1, 2, 3, 4, 5, 6, 7, 8, 9, 10, 11, 12, 13, 14, 15);
        acc = __builtin_amdgcn_wmma_f32_16x16x32_bf16(
            /*neg_a=*/false, a, /*neg_b=*/false, bm,
            /*c_mod=*/(short)0, acc, /*reuse_a=*/false, /*reuse_b=*/false);
    }
#pragma unroll
    for (int r = 0; r < 8; ++r) {
        const int m = m0 + r + half * 8;
        hproj[(size_t)m * HID + h0 + l16] = acc[r];
    }
}

// ---------------------------------------------------------------------------
// Kernel 4: sequential decode. 16 blocks (one per batch element, one WGP
// each), 256 threads = 8 wave32. All per-step state in LDS; weights stay
// L2-resident (shared/broadcast by all 16 WGPs).
// ---------------------------------------------------------------------------
__global__ void __launch_bounds__(256)
sla_decode_kernel(const float* __restrict__ featT,   // [B][N][C]
                  const float* __restrict__ hproj,   // [B][N][HID]
                  const int*   __restrict__ structure, // [B][T]
                  const float* __restrict__ h2h_w, const float* __restrict__ h2h_b,
                  const float* __restrict__ score_w,
                  const float* __restrict__ gru_wih, const float* __restrict__ gru_bih,
                  const float* __restrict__ gru_whh, const float* __restrict__ gru_bhh,
                  const float* __restrict__ sg1_w, const float* __restrict__ sg1_b,
                  const float* __restrict__ sg2_w, const float* __restrict__ sg2_b,
                  const float* __restrict__ lg1_w, const float* __restrict__ lg1_b,
                  const float* __restrict__ lg2_w, const float* __restrict__ lg2_b,
                  float* __restrict__ out)           // [B][T][38]
{
    const int b    = blockIdx.x;
    const int tid  = threadIdx.x;     // 0..255
    const int lane = tid & 31;
    const int wave = tid >> 5;        // 0..7

    __shared__ float s_hidden[HID];
    __shared__ float s_prev[HID];     // prev_proj; reused as lg1 hidden
    __shared__ float s_e[NPOS];       // logits -> alpha
    __shared__ float s_x[XDIM + 2];   // [context(512) | onehot(30)]
    __shared__ float s_red[256];
    __shared__ float s_hnew[HID];
    __shared__ float s_score[HID];
    __shared__ float s_t1[HID];       // sg1 hidden

    s_hidden[tid] = 0.f;
    s_score[tid]  = score_w[tid];
    __syncthreads();

    const float* fT = featT + (size_t)b * NPOS * CCH;
    const float* hp = hproj + (size_t)b * NPOS * HID;

    for (int t = 0; t < TSTEPS; ++t) {
        // warm GRU weight rows in L0/L2 while attention runs (global_prefetch_b8)
        __builtin_prefetch(gru_wih + (size_t)tid * XDIM, 0, 1);
        __builtin_prefetch(gru_whh + (size_t)tid * HID, 0, 1);

        // ---- 1. prev_proj[j] = h2h_w[j,:] . hidden + h2h_b[j] -------------
        {
            const float* wr = h2h_w + (size_t)tid * HID;
            float acc = h2h_b[tid];
#pragma unroll 4
            for (int k = 0; k < HID; k += 4) {
                const float4 w4 = *(const float4*)(wr + k);
                acc += w4.x * s_hidden[k]     + w4.y * s_hidden[k + 1]
                     + w4.z * s_hidden[k + 2] + w4.w * s_hidden[k + 3];
            }
            s_prev[tid] = acc;
        }
        __syncthreads();

        // ---- 2. e[n] = score . tanh(h_proj[n,:] + prev_proj) --------------
        // 8 waves x 50 positions; lane handles 8 consecutive h (coalesced).
        for (int n = wave; n < NPOS; n += 8) {
            const float* row = hp + (size_t)n * HID + lane * 8;
            const float4 a0 = *(const float4*)(row);
            const float4 a1 = *(const float4*)(row + 4);
            const int h0 = lane * 8;
            float acc;
            acc  = tanhf(a0.x + s_prev[h0 + 0]) * s_score[h0 + 0];
            acc += tanhf(a0.y + s_prev[h0 + 1]) * s_score[h0 + 1];
            acc += tanhf(a0.z + s_prev[h0 + 2]) * s_score[h0 + 2];
            acc += tanhf(a0.w + s_prev[h0 + 3]) * s_score[h0 + 3];
            acc += tanhf(a1.x + s_prev[h0 + 4]) * s_score[h0 + 4];
            acc += tanhf(a1.y + s_prev[h0 + 5]) * s_score[h0 + 5];
            acc += tanhf(a1.z + s_prev[h0 + 6]) * s_score[h0 + 6];
            acc += tanhf(a1.w + s_prev[h0 + 7]) * s_score[h0 + 7];
#pragma unroll
            for (int ofs = 16; ofs > 0; ofs >>= 1)
                acc += __shfl_down(acc, ofs, 32);
            if (lane == 0) s_e[n] = acc;
        }
        __syncthreads();

        // ---- 3. softmax over N=400 ----------------------------------------
        {
            float mloc = -1e30f;
            for (int i = tid; i < NPOS; i += 256) mloc = fmaxf(mloc, s_e[i]);
            s_red[tid] = mloc;
            __syncthreads();
#pragma unroll
            for (int s = 128; s > 0; s >>= 1) {
                if (tid < s) s_red[tid] = fmaxf(s_red[tid], s_red[tid + s]);
                __syncthreads();
            }
            const float mx = s_red[0];
            __syncthreads();
            float sloc = 0.f;
            for (int i = tid; i < NPOS; i += 256) {
                const float v = __expf(s_e[i] - mx);
                s_e[i] = v;
                sloc += v;
            }
            s_red[tid] = sloc;
            __syncthreads();
#pragma unroll
            for (int s = 128; s > 0; s >>= 1) {
                if (tid < s) s_red[tid] += s_red[tid + s];
                __syncthreads();
            }
            const float inv = 1.f / s_red[0];
            __syncthreads();
            for (int i = tid; i < NPOS; i += 256) s_e[i] *= inv;
        }
        __syncthreads();

        // ---- 4. context[c] = sum_n alpha[n] * featT[n,c]  (coalesced) -----
        {
            float acc0 = 0.f, acc1 = 0.f;
            for (int n = 0; n < NPOS; ++n) {
                const float a = s_e[n];
                const float* fr = fT + (size_t)n * CCH;
                acc0 += a * fr[tid];
                acc1 += a * fr[256 + tid];
            }
            s_x[tid]       = acc0;
            s_x[256 + tid] = acc1;
        }
        const int ch = structure[b * TSTEPS + t];
        if (tid < NEMB) s_x[CCH + tid] = (tid == ch) ? 1.f : 0.f;
        __syncthreads();

        // ---- 5. GRU cell: thread j owns gate component j ------------------
        {
            const float* wr = gru_wih + (size_t)tid * XDIM;
            const float* wz = gru_wih + (size_t)(HID + tid) * XDIM;
            const float* wn = gru_wih + (size_t)(2 * HID + tid) * XDIM;
            float ir = gru_bih[tid];
            float iz = gru_bih[HID + tid];
            float in_ = gru_bih[2 * HID + tid];
#pragma unroll 2
            for (int k = 0; k < XDIM; k += 2) {   // 542 even
                const float x0 = s_x[k], x1 = s_x[k + 1];
                ir  += wr[k] * x0 + wr[k + 1] * x1;
                iz  += wz[k] * x0 + wz[k + 1] * x1;
                in_ += wn[k] * x0 + wn[k + 1] * x1;
            }
            const float* vr = gru_whh + (size_t)tid * HID;
            const float* vz = gru_whh + (size_t)(HID + tid) * HID;
            const float* vn = gru_whh + (size_t)(2 * HID + tid) * HID;
            float hr = gru_bhh[tid];
            float hz = gru_bhh[HID + tid];
            float hn = gru_bhh[2 * HID + tid];
#pragma unroll 2
            for (int k = 0; k < HID; k += 2) {
                const float h0v = s_hidden[k], h1v = s_hidden[k + 1];
                hr += vr[k] * h0v + vr[k + 1] * h1v;
                hz += vz[k] * h0v + vz[k + 1] * h1v;
                hn += vn[k] * h0v + vn[k + 1] * h1v;
            }
            const float r  = 1.f / (1.f + __expf(-(ir + hr)));
            const float z  = 1.f / (1.f + __expf(-(iz + hz)));
            const float nn = tanhf(in_ + r * hn);
            s_hnew[tid] = (1.f - z) * nn + z * s_hidden[tid];
        }
        __syncthreads();
        s_hidden[tid] = s_hnew[tid];   // next-step read is after loop barrier

        // ---- 6. structure / loc generators --------------------------------
        {
            const float* w1 = sg1_w + (size_t)tid * HID;
            const float* w2 = lg1_w + (size_t)tid * HID;
            float a1 = sg1_b[tid];
            float a2 = lg1_b[tid];
#pragma unroll 2
            for (int k = 0; k < HID; k += 2) {
                const float h0v = s_hnew[k], h1v = s_hnew[k + 1];
                a1 += w1[k] * h0v + w1[k + 1] * h1v;
                a2 += w2[k] * h0v + w2[k + 1] * h1v;
            }
            s_t1[tid]   = a1;
            s_prev[tid] = a2;   // reuse s_prev as lg1 hidden
        }
        __syncthreads();

        float* orow = out + ((size_t)b * TSTEPS + t) * OUTW;
        if (tid < NEMB) {                       // wave 0: structure head
            const float* w = sg2_w + (size_t)tid * HID;
            float a = sg2_b[tid];
            for (int k = 0; k < HID; k += 4) {
                const float4 w4 = *(const float4*)(w + k);
                a += w4.x * s_t1[k]     + w4.y * s_t1[k + 1]
                   + w4.z * s_t1[k + 2] + w4.w * s_t1[k + 3];
            }
            orow[tid] = a;
        } else if (tid >= 64 && tid < 64 + LOCN) {  // wave 2: loc head
            const int o = tid - 64;
            const float* w = lg2_w + (size_t)o * HID;
            float a = lg2_b[o];
            for (int k = 0; k < HID; k += 4) {
                const float4 w4 = *(const float4*)(w + k);
                a += w4.x * s_prev[k]     + w4.y * s_prev[k + 1]
                   + w4.z * s_prev[k + 2] + w4.w * s_prev[k + 3];
            }
            orow[NEMB + o] = 1.f / (1.f + __expf(-a));
        }
        __syncthreads();
    }
}

// ---------------------------------------------------------------------------
extern "C" void kernel_launch(void* const* d_in, const int* in_sizes, int n_in,
                              void* d_out, int out_size, void* d_ws, size_t ws_size,
                              hipStream_t stream) {
    const float* fea       = (const float*)d_in[0];
    const int*   structure = (const int*)  d_in[1];
    const float* i2h_w     = (const float*)d_in[2];
    const float* h2h_w     = (const float*)d_in[3];
    const float* h2h_b     = (const float*)d_in[4];
    const float* score_w   = (const float*)d_in[5];
    const float* gru_wih   = (const float*)d_in[6];
    const float* gru_bih   = (const float*)d_in[7];
    const float* gru_whh   = (const float*)d_in[8];
    const float* gru_bhh   = (const float*)d_in[9];
    const float* sg1_w     = (const float*)d_in[10];
    const float* sg1_b     = (const float*)d_in[11];
    const float* sg2_w     = (const float*)d_in[12];
    const float* sg2_b     = (const float*)d_in[13];
    const float* lg1_w     = (const float*)d_in[14];
    const float* lg1_b     = (const float*)d_in[15];
    const float* lg2_w     = (const float*)d_in[16];
    const float* lg2_b     = (const float*)d_in[17];
    float* out = (float*)d_out;

    // workspace layout (bytes):
    //   featT f32 [16][400][512]  = 13,107,200
    //   hproj f32 [16][400][256]  =  6,553,600
    //   featB bf16 [16][400][512] =  6,553,600
    //   i2hB  bf16 [256][512]     =    262,144     total ~26.5 MB
    float*  featT = (float*)d_ws;
    float*  hproj = featT + (size_t)BSZ * NPOS * CCH;
    __bf16* featB = (__bf16*)(hproj + (size_t)BSZ * NPOS * HID);
    __bf16* i2hB  = featB + (size_t)BSZ * NPOS * CCH;

    sla_transpose_kernel<<<dim3(NPOS / 16, CCH / 16, BSZ), 256, 0, stream>>>(
        fea, featT, featB);
    sla_cvt_w_kernel<<<(HID * CCH) / (256 * 4), 256, 0, stream>>>(i2h_w, i2hB);
    sla_hproj_wmma_kernel<<<dim3((BSZ * NPOS) / 16, (HID / 16) / 4), 128, 0, stream>>>(
        featB, i2hB, hproj);
    sla_decode_kernel<<<BSZ, 256, 0, stream>>>(
        featT, hproj, structure,
        h2h_w, h2h_b, score_w,
        gru_wih, gru_bih, gru_whh, gru_bhh,
        sg1_w, sg1_b, sg2_w, sg2_b,
        lg1_w, lg1_b, lg2_w, lg2_b, out);
}
